// BiNet_25615184953554
// MI455X (gfx1250) — matrix operand order repaired and checked by reading, not compiled, not run on hardware
//
#include <hip/hip_runtime.h>
#include <hip/hip_bf16.h>
#include <math.h>
#include <stdint.h>

typedef _Float16 f16;
typedef __attribute__((ext_vector_type(8)))  _Float16 v8h;
typedef __attribute__((ext_vector_type(16))) _Float16 v16h;
typedef __attribute__((ext_vector_type(8)))  float    v8f;

#define WMMA16x16x32(a,b,c) \
  __builtin_amdgcn_wmma_f32_16x16x32_f16(false,(a),false,(b),(short)0,(c),false,false)

#define GEMM_BT   1
#define GEMM_RELU 2

// ---------------------------------------------------------------- f32 -> f16
__global__ void cvt_f32_to_f16(const float* __restrict__ x, f16* __restrict__ y,
                               long long n) {
  long long i = (long long)blockIdx.x * blockDim.x + threadIdx.x;
  long long stride = (long long)gridDim.x * blockDim.x;
  for (; i < n; i += stride) y[i] = (f16)x[i];
}

// ---------------------------------------------------------------- WMMA GEMM
// C[b] = A[b] (MxK f16, row-major) * op(B[b]) + bias, f32 out.
// flags&GEMM_BT: B is NxK row-major (C = A*B^T).  else B is KxN row-major.
// Block = 256 threads = 8 waves (2 M-waves x 4 N-waves) -> 32x64 tile, BK=32.
// Interior blocks: double-buffered LDS filled with GLOBAL_LOAD_ASYNC_TO_LDS_B128
// (ASYNCcnt), overlapping next-tile DMA with current-tile WMMA.
__global__ __launch_bounds__(256)
void wmma_gemm(const f16* __restrict__ A, int lda, long long sA,
               const f16* __restrict__ B, int ldb, long long sB,
               float* __restrict__ C, int ldc, long long sC,
               int M, int N, int K, const float* __restrict__ bias, int flags)
{
  __shared__ __align__(16) f16 As[2][32][40];    // As[buf][m][k], 16B row pad
  __shared__ __align__(16) f16 BsT[2][64][40];   // BsT[buf][n][k], K-contiguous
  const int tid = threadIdx.x;
  const int m0 = blockIdx.y * 32, n0 = blockIdx.x * 64;
  const f16* Ab = A + (long long)blockIdx.z * sA;
  const f16* Bb = B + (long long)blockIdx.z * sB;
  float*     Cb = C + (long long)blockIdx.z * sC;
  const int w = tid >> 5, lane = tid & 31, hf = lane >> 4, ln = lane & 15;
  const int wm = w >> 2, wn = w & 3;
  v8f acc = {};

  const bool fast = (m0 + 32 <= M) && (n0 + 64 <= N) && ((K & 31) == 0);

  if (fast) {
    // LDS byte addresses = low 32 bits of flat shared pointers (ISA 10.2).
    const unsigned aBase = (unsigned)(unsigned long long)(size_t)&As[0][0][0];
    const unsigned bBase = (unsigned)(unsigned long long)(size_t)&BsT[0][0][0];
    const unsigned aBuf = 32u * 40u * 2u, bBuf = 64u * 40u * 2u;

    auto stage = [&](int kk, int buf) {
      // A tile 32x32 f16 = 128 x 16B : threads 0..127
      if (tid < 128) {
        int m = tid >> 2, k = (tid & 3) * 8;
        unsigned lo = aBase + (unsigned)buf * aBuf + (unsigned)((m * 40 + k) * 2);
        const f16* g = Ab + (long long)(m0 + m) * lda + kk + k;
        asm volatile("global_load_async_to_lds_b128 %0, %1, off"
                     :: "v"(lo), "v"(g) : "memory");
      }
      if (flags & GEMM_BT) {
        // B tile 64x32 f16 = 256 x 16B : all threads, straight copy
        int n = tid >> 2, k = (tid & 3) * 8;
        unsigned lo = bBase + (unsigned)buf * bBuf + (unsigned)((n * 40 + k) * 2);
        const f16* g = Bb + (long long)(n0 + n) * ldb + kk + k;
        asm volatile("global_load_async_to_lds_b128 %0, %1, off"
                     :: "v"(lo), "v"(g) : "memory");
      } else {
        // KxN source needs a transpose: regular ds stores (DScnt)
#pragma unroll
        for (int i = 0; i < 8; i++) {
          int e = tid + i * 256, k = e >> 6, n = e & 63;
          BsT[buf][n][k] = Bb[(long long)(kk + k) * ldb + n0 + n];
        }
      }
    };

    int cur = 0;
    stage(0, 0);
    for (int kk = 0; kk < K; kk += 32) {
      asm volatile("s_wait_asynccnt 0" ::: "memory");
      __syncthreads();                        // tile[cur] visible to all waves
      if (kk + 32 < K) stage(kk + 32, cur ^ 1);  // overlap next-tile DMA

      union { v16h v; v8h h[2]; } af, bfr;
      af.h[0] = *(const v8h*)&As[cur][wm * 16 + ln][16 * hf];
      af.h[1] = *(const v8h*)&As[cur][wm * 16 + ln][16 * hf + 8];
      bfr.h[0] = *(const v8h*)&BsT[cur][wn * 16 + ln][16 * hf];
      bfr.h[1] = *(const v8h*)&BsT[cur][wn * 16 + ln][16 * hf + 8];
      acc = WMMA16x16x32(af.v, bfr.v, acc);
      cur ^= 1;
    }
  } else {
    // -------- guarded fallback (edge tiles: FC head M=16 / N=5) --------
    for (int kk = 0; kk < K; kk += 32) {
#pragma unroll
      for (int i = 0; i < 4; i++) {
        int e = tid + i * 256, m = e >> 5, k = e & 31;
        int gm = m0 + m, gk = kk + k;
        As[0][m][k] = (gm < M && gk < K) ? Ab[(long long)gm * lda + gk] : (f16)0.f;
      }
      if (flags & GEMM_BT) {
#pragma unroll
        for (int i = 0; i < 8; i++) {
          int e = tid + i * 256, n = e >> 5, k = e & 31;
          int gn = n0 + n, gk = kk + k;
          BsT[0][n][k] = (gn < N && gk < K) ? Bb[(long long)gn * ldb + gk] : (f16)0.f;
        }
      } else {
#pragma unroll
        for (int i = 0; i < 8; i++) {
          int e = tid + i * 256, k = e >> 6, n = e & 63;
          int gn = n0 + n, gk = kk + k;
          BsT[0][n][k] = (gn < N && gk < K) ? Bb[(long long)gk * ldb + gn] : (f16)0.f;
        }
      }
      if (kk + 32 < K)
        __builtin_prefetch(&Ab[(long long)(m0 + lane) * lda + kk + 32], 0, 1);
      __syncthreads();

      union { v16h v; v8h h[2]; } af, bfr;
      af.h[0] = *(const v8h*)&As[0][wm * 16 + ln][16 * hf];
      af.h[1] = *(const v8h*)&As[0][wm * 16 + ln][16 * hf + 8];
      bfr.h[0] = *(const v8h*)&BsT[0][wn * 16 + ln][16 * hf];
      bfr.h[1] = *(const v8h*)&BsT[0][wn * 16 + ln][16 * hf + 8];
      acc = WMMA16x16x32(af.v, bfr.v, acc);
      __syncthreads();
    }
  }

#pragma unroll
  for (int v = 0; v < 8; v++) {               // C/D: m = v + 8*hf, n = ln
    int gm = m0 + wm * 16 + v + 8 * hf;
    int gn = n0 + wn * 16 + ln;
    if (gm < M && gn < N) {
      float r = acc[v] + (bias ? bias[gn] : 0.f);
      if (flags & GEMM_RELU) r = fmaxf(r, 0.f);
      Cb[(long long)gm * ldc + gn] = r;
    }
  }
}

// ------------------------------------------------- softmax over axis=1 (q) + f16
__global__ void softmax_axis1(const float* __restrict__ S, f16* __restrict__ P,
                              int batch, int Q, int Kc)
{
  int idx = blockIdx.x * blockDim.x + threadIdx.x;
  if (idx >= batch * Kc) return;
  int b = idx / Kc, kc = idx - b * Kc;
  const float* s = S + (long long)b * Q * Kc + kc;
  f16* p = P + (long long)b * Q * Kc + kc;
  float mx = -3.4e38f;
  for (int q = 0; q < Q; q++) mx = fmaxf(mx, s[(long long)q * Kc]);
  float sum = 0.f;
  for (int q = 0; q < Q; q++) sum += __expf(s[(long long)q * Kc] - mx);
  float inv = 1.f / sum;
  for (int q = 0; q < Q; q++)
    p[(long long)q * Kc] = (f16)(__expf(s[(long long)q * Kc] - mx) * inv);
}

// ---------------------------------------------------------- fused BiLSTM scan
// One block per (direction, sequence). 512 threads = 16 waves; wave w owns
// hidden columns [w*32, w*32+32). h kept in LDS (f16), c in C-frag registers.
// Per step: g = xs[t] + h @ whh^T via WMMA (K=512), then gate nonlinearities.
struct LstmArgs {
  const float* xsF[8];
  const float* xsB[8];
  float* out[8];   // (B,T,1024), fwd writes cols 0..511, bwd cols 512..1023
  int T[8];
};

__global__ __launch_bounds__(512)
void lstm_seq(LstmArgs args, const f16* __restrict__ whhF,
              const f16* __restrict__ whhB)
{
  const int dir = blockIdx.x, seq = blockIdx.y;
  const int T = args.T[seq];
  const float* xs = dir ? args.xsB[seq] : args.xsF[seq];
  const f16* whh = dir ? whhB : whhF;     // (2048 x 512) row-major, f16
  float* out = args.out[seq] + dir * 512;

  __shared__ __align__(16) f16 hsh[16][520];  // h: 16 batch x 512, padded
  const int tid = threadIdx.x, w = tid >> 5, lane = tid & 31;
  const int hf = lane >> 4, ln = lane & 15;
  const int jbase = w * 32;

  for (int i = tid; i < 16 * 520; i += 512) (&hsh[0][0])[i] = (f16)0.f;
  v8f c0 = {}, c1 = {};
  __syncthreads();

  for (int step = 0; step < T; ++step) {
    const int t = dir ? (T - 1 - step) : step;
    v8f zero = {};
    v8f acc[2][4];
#pragma unroll
    for (int jt = 0; jt < 2; jt++)
#pragma unroll
      for (int g = 0; g < 4; g++) acc[jt][g] = zero;

    for (int kk = 0; kk < 512; kk += 32) {
      union { v16h v; v8h h[2]; } af;     // A = h (16 x 512) from LDS
      af.h[0] = *(const v8h*)&hsh[ln][kk + 16 * hf];
      af.h[1] = *(const v8h*)&hsh[ln][kk + 16 * hf + 8];
#pragma unroll
      for (int jt = 0; jt < 2; jt++) {
#pragma unroll
        for (int g = 0; g < 4; g++) {
          int n = g * 512 + jbase + jt * 16 + ln;   // whh row = output gate col
          v16h bfrag = *(const v16h*)(whh + (long long)n * 512 + kk + 16 * hf);
          acc[jt][g] = WMMA16x16x32(af.v, bfrag, acc[jt][g]);
        }
      }
    }
    __syncthreads();                       // all h reads done before overwrite

#pragma unroll
    for (int jt = 0; jt < 2; jt++) {
      int j = jbase + jt * 16 + ln;
#pragma unroll
      for (int v = 0; v < 8; v++) {
        int m = v + 8 * hf;                // batch row
        long long xrow = ((long long)m * T + t) * 2048;
        float gi = acc[jt][0][v] + xs[xrow + j];
        float gf = acc[jt][1][v] + xs[xrow + 512 + j];
        float gg = acc[jt][2][v] + xs[xrow + 1024 + j];
        float go = acc[jt][3][v] + xs[xrow + 1536 + j];
        float si = 1.f / (1.f + __expf(-gi));
        float sf = 1.f / (1.f + __expf(-gf));
        float so = 1.f / (1.f + __expf(-go));
        float cp = jt ? c1[v] : c0[v];
        float cv = sf * cp + si * tanhf(gg);
        if (jt) c1[v] = cv; else c0[v] = cv;
        float hv = so * tanhf(cv);
        hsh[m][j] = (f16)hv;
        out[((long long)m * T + t) * 1024 + j] = hv;
      }
    }
    __syncthreads();
  }
}

// ------------------------------------------------------------- feats gather
struct GatherArgs { const float* src[8]; int T[8]; };

__global__ void gather_feats(GatherArgs ga, f16* __restrict__ feats) {
  int idx = blockIdx.x * blockDim.x + threadIdx.x;
  if (idx >= 16 * 20480) return;
  int b = idx / 20480, col = idx - b * 20480;
  int s = col >> 12, r = col & 4095, which = r >> 10, j = r & 1023;
  int si = (which == 3) ? 3 + s : which;   // {keytext, self_text, self_vis, sq[s]}
  feats[idx] = (f16)ga.src[si][(long long)b * ga.T[si] * 1024 + j];
}

// =========================================================================
extern "C" void kernel_launch(void* const* d_in, const int* in_sizes, int n_in,
                              void* d_out, int out_size, void* d_ws, size_t ws_size,
                              hipStream_t stream)
{
  (void)in_sizes; (void)n_in; (void)out_size; (void)ws_size;
  const int Ls[8] = {64, 64, 64, 64, 64, 256, 128, 256};

  // pytree flatten order: bf0..bf7, then params dict sorted by key:
  // att.* [8..13] and attw.* [14..19] are DEAD (never reach feats) -> skipped.
  const float* bf[8]; for (int i = 0; i < 8; i++) bf[i] = (const float*)d_in[i];
  const float* fc_b1   = (const float*)d_in[20];
  const float* fc_b2   = (const float*)d_in[21];
  const float* fc_w1   = (const float*)d_in[22];
  const float* fc_w2   = (const float*)d_in[23];
  const float* il_bb   = (const float*)d_in[24];
  const float* il_bf_  = (const float*)d_in[25];
  const float* il_whhb = (const float*)d_in[26];
  const float* il_whhf = (const float*)d_in[27];
  const float* il_wihb = (const float*)d_in[28];
  const float* il_wihf = (const float*)d_in[29];
  const float* rw_bb   = (const float*)d_in[30];
  const float* rw_bf_  = (const float*)d_in[31];
  const float* rw_whhb = (const float*)d_in[32];
  const float* rw_whhf = (const float*)d_in[33];
  const float* rw_wihb = (const float*)d_in[34];
  const float* rw_wihf = (const float*)d_in[35];

  size_t off = 0;
  auto alloc = [&](size_t bytes) -> void* {
    void* p = (char*)d_ws + off;
    off += (bytes + 255) & ~(size_t)255;
    return p;
  };
  auto CVT = [&](const float* x, f16* y, long long n) {
    int blocks = (int)((n + 2047) / 2048); if (blocks > 4096) blocks = 4096;
    if (blocks < 1) blocks = 1;
    cvt_f32_to_f16<<<blocks, 256, 0, stream>>>(x, y, n);
  };
  auto GEMM = [&](const f16* A, int lda, long long sA,
                  const f16* B, int ldb, long long sB,
                  float* C, int ldc, long long sC,
                  int M, int N, int K, const float* bias, int flags, int batch) {
    dim3 g((N + 63) / 64, (M + 31) / 32, batch);
    wmma_gemm<<<g, 256, 0, stream>>>(A, lda, sA, B, ldb, sB, C, ldc, sC,
                                     M, N, K, bias, flags);
  };

  // ---- f16 weights
  f16* wihIl[2]; f16* whhIl[2]; f16* wihRw[2]; f16* whhRw[2];
  wihIl[0] = (f16*)alloc(2048LL * 768 * 2);  wihIl[1] = (f16*)alloc(2048LL * 768 * 2);
  whhIl[0] = (f16*)alloc(2048LL * 512 * 2);  whhIl[1] = (f16*)alloc(2048LL * 512 * 2);
  wihRw[0] = (f16*)alloc(2048LL * 1024 * 2); wihRw[1] = (f16*)alloc(2048LL * 1024 * 2);
  whhRw[0] = (f16*)alloc(2048LL * 512 * 2);  whhRw[1] = (f16*)alloc(2048LL * 512 * 2);
  f16* w1h = (f16*)alloc(2048LL * 20480 * 2);
  f16* w2h = (f16*)alloc(5LL * 2048 * 2);
  CVT(il_wihf, wihIl[0], 2048LL * 768);  CVT(il_wihb, wihIl[1], 2048LL * 768);
  CVT(il_whhf, whhIl[0], 2048LL * 512);  CVT(il_whhb, whhIl[1], 2048LL * 512);
  CVT(rw_wihf, wihRw[0], 2048LL * 1024); CVT(rw_wihb, wihRw[1], 2048LL * 1024);
  CVT(rw_whhf, whhRw[0], 2048LL * 512);  CVT(rw_whhb, whhRw[1], 2048LL * 512);
  CVT(fc_w1, w1h, 2048LL * 20480);       CVT(fc_w2, w2h, 5LL * 2048);

  // ---- activations
  f16* bfh[8]; float* xsF[8]; float* xsB[8]; float* outs[8]; f16* outs16[8];
  for (int i = 0; i < 8; i++) {
    long long rows = 16LL * Ls[i];
    bfh[i]    = (f16*)  alloc(rows * 768 * 2);
    xsF[i]    = (float*)alloc(rows * 2048 * 4);
    xsB[i]    = (float*)alloc(rows * 2048 * 4);
    outs[i]   = (float*)alloc(rows * 1024 * 4);
    outs16[i] = (f16*)  alloc(rows * 1024 * 2);
    CVT(bf[i], bfh[i], rows * 768);
  }

  // ---- stage 1: il BiLSTM (input proj as big WMMA GEMMs, then fused scans)
  for (int i = 0; i < 8; i++) {
    int Mr = 16 * Ls[i];
    GEMM(bfh[i], 768, 0, wihIl[0], 768, 0, xsF[i], 2048, 0,
         Mr, 2048, 768, il_bf_, GEMM_BT, 1);
    GEMM(bfh[i], 768, 0, wihIl[1], 768, 0, xsB[i], 2048, 0,
         Mr, 2048, 768, il_bb, GEMM_BT, 1);
  }
  {
    LstmArgs la;
    for (int i = 0; i < 8; i++) {
      la.xsF[i] = xsF[i]; la.xsB[i] = xsB[i]; la.out[i] = outs[i]; la.T[i] = Ls[i];
    }
    lstm_seq<<<dim3(2, 8), 512, 0, stream>>>(la, whhIl[0], whhIl[1]);
  }
  for (int i = 0; i < 8; i++) CVT(outs[i], outs16[i], 16LL * Ls[i] * 1024);

  // ---- stage 2: soft attentions (only the 8 live ones)
  const int Ta[8] = {256, 256, 128, 64, 64, 64, 64, 64};
  const int qi[8] = {7, 5, 6, 0, 1, 2, 3, 4};
  const int ki[8] = {5, 5, 6, 0, 1, 2, 3, 4};
  float* Sbuf = (float*)alloc(16LL * 256 * 256 * 4);
  f16*   S16  = (f16*)  alloc(16LL * 256 * 256 * 2);
  float* attout[8]; f16* attout16[8];
  for (int a = 0; a < 8; a++) {
    attout[a]   = (float*)alloc(16LL * Ta[a] * 1024 * 4);
    attout16[a] = (f16*)  alloc(16LL * Ta[a] * 1024 * 2);
  }
  for (int a = 0; a < 8; a++) {
    int Q = Ls[qi[a]], Kc = Ls[ki[a]];
    // S = q @ k^T  (batched NT)
    GEMM(outs16[qi[a]], 1024, (long long)Q * 1024,
         outs16[ki[a]], 1024, (long long)Kc * 1024,
         Sbuf, Kc, (long long)Q * Kc, Q, Kc, 1024, nullptr, GEMM_BT, 16);
    int tot = 16 * Kc;
    softmax_axis1<<<(tot + 255) / 256, 256, 0, stream>>>(Sbuf, S16, 16, Q, Kc);
    // out = attn @ v  (batched NN)
    GEMM(S16, Kc, (long long)Q * Kc,
         outs16[ki[a]], 1024, (long long)Kc * 1024,
         attout[a], 1024, (long long)Q * 1024, Q, 1024, Kc, nullptr, 0, 16);
    CVT(attout[a], attout16[a], 16LL * Q * 1024);
  }

  // ---- stage 3: raw BiLSTM (reuse xs buffers; lengths match by this slot map)
  const int slot[8] = {5, 7, 6, 0, 1, 2, 3, 4};
  float* rawout[8];
  for (int a = 0; a < 8; a++) rawout[a] = (float*)alloc(16LL * Ta[a] * 1024 * 4);
  for (int a = 0; a < 8; a++) {
    int Mr = 16 * Ta[a];
    GEMM(attout16[a], 1024, 0, wihRw[0], 1024, 0, xsF[slot[a]], 2048, 0,
         Mr, 2048, 1024, rw_bf_, GEMM_BT, 1);
    GEMM(attout16[a], 1024, 0, wihRw[1], 1024, 0, xsB[slot[a]], 2048, 0,
         Mr, 2048, 1024, rw_bb, GEMM_BT, 1);
  }
  {
    LstmArgs lr;
    for (int a = 0; a < 8; a++) {
      lr.xsF[a] = xsF[slot[a]]; lr.xsB[a] = xsB[slot[a]];
      lr.out[a] = rawout[a]; lr.T[a] = Ta[a];
    }
    lstm_seq<<<dim3(2, 8), 512, 0, stream>>>(lr, whhRw[0], whhRw[1]);
  }

  // ---- stage 4: gather t=0 rows -> feats, then FC head
  f16* feats = (f16*)alloc(16LL * 20480 * 2);
  {
    GatherArgs ga;
    for (int a = 0; a < 8; a++) { ga.src[a] = rawout[a]; ga.T[a] = Ta[a]; }
    gather_feats<<<(16 * 20480 + 255) / 256, 256, 0, stream>>>(ga, feats);
  }
  float* h1  = (float*)alloc(16LL * 2048 * 4);
  f16*   h1h = (f16*)  alloc(16LL * 2048 * 2);
  GEMM(feats, 20480, 0, w1h, 20480, 0, h1, 2048, 0,
       16, 2048, 20480, fc_b1, GEMM_BT | GEMM_RELU, 1);
  CVT(h1, h1h, 16LL * 2048);
  GEMM(h1h, 2048, 0, w2h, 2048, 0, (float*)d_out, 5, 0,
       16, 5, 2048, fc_b2, GEMM_BT, 1);
}
